// mix_model_10385230922262
// MI455X (gfx1250) — matrix operand order repaired and checked by reading, not compiled, not run on hardware
//
#include <hip/hip_runtime.h>
#include <hip/hip_bf16.h>

// MI455X / gfx1250, wave32. D = A(16x32 f16) x B(32x16 f16) + C(16x16 f32)
typedef __attribute__((ext_vector_type(16))) _Float16 v16h;
typedef __attribute__((ext_vector_type(8)))  float    v8f;

#define B_   32
#define G_   128
#define T_   128
#define FIN_ 318

// ---- LDS layout (bytes), total 62400 < 64KB ----
#define OFF_W1A 0        // 50x320 f16  (K padded 318->320, zeros)
#define OFF_W1B 32000    // 20x64  f16  (K padded 50->64, zeros)
#define OFF_W1C 34560    // 20x32  f16  (K padded 20->32, zeros)
#define OFF_B1A 35840    // 64  f32 (zero padded)
#define OFF_B1B 36096    // 32  f32
#define OFF_B1C 36224    // 32  f32
#define OFF_PW1 36352    // 10x20 f32
#define OFF_PB1 37152    // 10 f32
#define OFF_PW2 37216    // 10 f32
#define OFF_PB2 37280    // 1 f32
#define OFF_SC  37296    // 128 f32 (attention scores / weights)
#define OFF_RED 37808    // reduction scratch
#define OFF_H1  37824    // 128x64 f16 h1; later aliased as 128x20 f32 h3
#define OFF_H2  54208    // 128x32 f16 h2
#define SMEM_BYTES 62400

static_assert(SMEM_BYTES <= 64 * 1024, "LDS budget");

__device__ __forceinline__ v8f wmma_f16(v16h a, v16h b, v8f c) {
  // v_wmma_f32_16x16x32_f16
  return __builtin_amdgcn_wmma_f32_16x16x32_f16(
      /*neg_a=*/false, a, /*neg_b=*/false, b,
      /*c_mod=*/(short)0, c, /*reuse_a=*/false, /*reuse_b=*/false);
}

// A-matrix 16x32 f16 lane layout: lane<16 holds row M=lane, K={k0..k0+7, k0+16..k0+23};
// lane>=16 holds row M=lane-16, K shifted by +8.
__device__ __forceinline__ v16h lds_load_A16(const _Float16* base, int pitch,
                                             int m, int k0, int lane) {
  const int kh = (lane >> 4) << 3;
  const _Float16* p = base + m * pitch + k0 + kh;
  v16h a;
#pragma unroll
  for (int i = 0; i < 8; ++i) { a[i] = p[i]; a[i + 8] = p[i + 16]; }
  return a;
}

// B-matrix 32x16 f16 lane layout: lane<16 holds col N=lane, K={k0..k0+7, k0+16..k0+23};
// lane>=16 holds col N=lane-16, K shifted by +8. Since B[k][n] = W[n][k], a B column
// is a contiguous K-run of a row-major W stored in LDS.
__device__ __forceinline__ v16h lds_load_B16(const _Float16* base, int pitch,
                                             int n, int k0, int lane) {
  const int kb = (lane >> 4) << 3;
  const _Float16* p = base + n * pitch + k0 + kb;
  v16h b;
#pragma unroll
  for (int i = 0; i < 8; ++i) { b[i] = p[i]; b[i + 8] = p[i + 16]; }
  return b;
}

__global__ __launch_bounds__(256) void fused_mlp_pool_t(
    const float* __restrict__ x,
    const float* __restrict__ W1a, const float* __restrict__ b1a,
    const float* __restrict__ W1b, const float* __restrict__ b1b,
    const float* __restrict__ W1c, const float* __restrict__ b1c,
    const float* __restrict__ pW1, const float* __restrict__ pb1,
    const float* __restrict__ pW2, const float* __restrict__ pb2,
    float* __restrict__ r1out)  // (B*G, 20)
{
  __shared__ __align__(32) unsigned char smem[SMEM_BYTES];
  _Float16* sW1a = (_Float16*)(smem + OFF_W1A);
  _Float16* sW1b = (_Float16*)(smem + OFF_W1B);
  _Float16* sW1c = (_Float16*)(smem + OFF_W1C);
  float* sB1a = (float*)(smem + OFF_B1A);
  float* sB1b = (float*)(smem + OFF_B1B);
  float* sB1c = (float*)(smem + OFF_B1C);
  float* sPW1 = (float*)(smem + OFF_PW1);
  float* sPB1 = (float*)(smem + OFF_PB1);
  float* sPW2 = (float*)(smem + OFF_PW2);
  float* sPB2 = (float*)(smem + OFF_PB2);
  float* sSC  = (float*)(smem + OFF_SC);
  float* sRED = (float*)(smem + OFF_RED);
  _Float16* sH1 = (_Float16*)(smem + OFF_H1);
  _Float16* sH2 = (_Float16*)(smem + OFF_H2);
  float* sH3 = (float*)(smem + OFF_H1);  // alias: h1 is dead after layer 2

  const int tid  = threadIdx.x;
  const int lane = tid & 31;
  const int wave = tid >> 5;   // 8 waves; each owns M-tile = wave (16 rows of T)
  const int bg   = blockIdx.x; // (b*G + g)

  // ---- Stage weights into LDS (f16, zero padded) -- L2-resident in global ----
  for (int i = tid; i < 50 * 320; i += 256) {
    int k = i % 320;
    sW1a[i] = (_Float16)((k < FIN_) ? W1a[(i / 320) * FIN_ + k] : 0.f);
  }
  for (int i = tid; i < 20 * 64; i += 256) {
    int k = i % 64;
    sW1b[i] = (_Float16)((k < 50) ? W1b[(i / 64) * 50 + k] : 0.f);
  }
  for (int i = tid; i < 20 * 32; i += 256) {
    int k = i % 32;
    sW1c[i] = (_Float16)((k < 20) ? W1c[(i / 32) * 20 + k] : 0.f);
  }
  if (tid < 64) sB1a[tid] = (tid < 50) ? b1a[tid] : 0.f;
  if (tid < 32) sB1b[tid] = (tid < 20) ? b1b[tid] : 0.f;
  if (tid < 32) sB1c[tid] = (tid < 20) ? b1c[tid] : 0.f;
  if (tid < 200) sPW1[tid] = pW1[tid];
  if (tid < 10)  sPB1[tid] = pb1[tid];
  if (tid < 10)  sPW2[tid] = pW2[tid];
  if (tid == 0)  sPB2[0]   = pb2[0];
  __syncthreads();

  const int mrow  = wave * 16 + (lane & 15);          // row this lane feeds A with
  const int mbase = wave * 16 + ((lane >> 4) << 3);   // D-fragment row base
  const int ncb   = (lane & 15);                      // D-fragment col within tile

  // ============ Layer 1: [128x318] @ [318x50]^T via f16 WMMA ============
  // A streamed straight from global fp32 (the only large memory stream).
  {
    const float* xrow = x + ((size_t)bg * T_ + mrow) * FIN_;
    const int kh = (lane >> 4) << 3;
    v8f acc[4];
#pragma unroll
    for (int nt = 0; nt < 4; ++nt)
#pragma unroll
      for (int r = 0; r < 8; ++r) acc[nt][r] = 0.f;

    for (int kt = 0; kt < 10; ++kt) {
      const int k0 = kt * 32;
      v16h a;
      if (kt < 9) {  // fully in-range: contiguous 8+8 float runs, f32 -> f16
#pragma unroll
        for (int i = 0; i < 8; ++i) {
          a[i]     = (_Float16)xrow[k0 + kh + i];
          a[i + 8] = (_Float16)xrow[k0 + kh + 16 + i];
        }
      } else {       // K tail 288..319, guard against 318
#pragma unroll
        for (int i = 0; i < 8; ++i) {
          int ka = k0 + kh + i, kb = ka + 16;
          a[i]     = (_Float16)((ka < FIN_) ? xrow[ka] : 0.f);
          a[i + 8] = (_Float16)((kb < FIN_) ? xrow[kb] : 0.f);
        }
      }
#pragma unroll
      for (int nt = 0; nt < 4; ++nt) {
        v16h bf = lds_load_B16(sW1a, 320, nt * 16 + ncb, k0, lane);
        acc[nt] = wmma_f16(a, bf, acc[nt]);
      }
    }
    // h1 = relu(acc + b1a); zero pad columns >= 50 (avoid NaN feed into layer 2)
#pragma unroll
    for (int nt = 0; nt < 4; ++nt) {
      const int ncol = nt * 16 + ncb;
      const float bias = sB1a[ncol];
      const bool live = (ncol < 50);
#pragma unroll
      for (int r = 0; r < 8; ++r) {
        float v = live ? fmaxf(acc[nt][r] + bias, 0.f) : 0.f;
        sH1[(mbase + r) * 64 + ncol] = (_Float16)v;
      }
    }
  }
  __syncthreads();

  // ============ Layer 2: [128x50] @ [50x20]^T ============
  {
    v8f acc[2];
#pragma unroll
    for (int nt = 0; nt < 2; ++nt)
#pragma unroll
      for (int r = 0; r < 8; ++r) acc[nt][r] = 0.f;
#pragma unroll
    for (int kt = 0; kt < 2; ++kt) {
      const int k0 = kt * 32;
      v16h a = lds_load_A16(sH1, 64, mrow, k0, lane);
#pragma unroll
      for (int nt = 0; nt < 2; ++nt) {
        v16h bf = lds_load_B16(sW1b, 64, nt * 16 + ncb, k0, lane);
        acc[nt] = wmma_f16(a, bf, acc[nt]);
      }
    }
#pragma unroll
    for (int nt = 0; nt < 2; ++nt) {
      const int ncol = nt * 16 + ncb;
      const float bias = (ncol < 32) ? sB1b[ncol] : 0.f;
      const bool live = (ncol < 20);
#pragma unroll
      for (int r = 0; r < 8; ++r) {
        float v = live ? fmaxf(acc[nt][r] + bias, 0.f) : 0.f;
        sH2[(mbase + r) * 32 + ncol] = (_Float16)v;
      }
    }
  }
  __syncthreads();

  // ============ Layer 3: [128x20] @ [20x20]^T -> h3 (f32, aliases h1) ============
  {
    v8f acc[2];
#pragma unroll
    for (int nt = 0; nt < 2; ++nt)
#pragma unroll
      for (int r = 0; r < 8; ++r) acc[nt][r] = 0.f;
    v16h a = lds_load_A16(sH2, 32, mrow, 0, lane);
#pragma unroll
    for (int nt = 0; nt < 2; ++nt) {
      v16h bf = lds_load_B16(sW1c, 32, nt * 16 + ncb, 0, lane);
      acc[nt] = wmma_f16(a, bf, acc[nt]);
    }
#pragma unroll
    for (int nt = 0; nt < 2; ++nt) {
      const int ncol = nt * 16 + ncb;
      if (ncol < 20) {
        const float bias = sB1c[ncol];
#pragma unroll
        for (int r = 0; r < 8; ++r)
          sH3[(mbase + r) * 20 + ncol] = fmaxf(acc[nt][r] + bias, 0.f);
      }
    }
  }
  __syncthreads();

  // ============ Self-attention pool over T (tiny: VALU) ============
  if (tid < T_) {
    const float* h = sH3 + tid * 20;
    float s = sPB2[0];
#pragma unroll
    for (int j = 0; j < 10; ++j) {
      float u = sPB1[j];
#pragma unroll
      for (int k = 0; k < 20; ++k) u += sPW1[j * 20 + k] * h[k];
      s += sPW2[j] * fmaxf(u, 0.f);
    }
    sSC[tid] = s;
  }
  __syncthreads();
  if (tid == 0) {
    float mx = -1e30f;
    for (int t = 0; t < T_; ++t) mx = fmaxf(mx, sSC[t]);
    float sum = 0.f;
    for (int t = 0; t < T_; ++t) { float e = __expf(sSC[t] - mx); sSC[t] = e; sum += e; }
    sRED[0] = 1.f / sum;
  }
  __syncthreads();
  if (tid < 20) {
    const float inv = sRED[0];
    float acc = 0.f;
    for (int t = 0; t < T_; ++t) acc += sH3[t * 20 + tid] * sSC[t];
    r1out[(size_t)bg * 20 + tid] = acc * inv;
  }
}

// ============ Pool over G + L2-normalize + 20->10->1 head ============
__global__ __launch_bounds__(128) void pool_g_head(
    const float* __restrict__ r1,
    const float* __restrict__ qW1, const float* __restrict__ qb1,
    const float* __restrict__ qW2, const float* __restrict__ qb2,
    const float* __restrict__ W3a, const float* __restrict__ b3a,
    const float* __restrict__ W3b, const float* __restrict__ b3b,
    float* __restrict__ out)
{
  __shared__ float sS[G_];
  __shared__ float sR[20];
  __shared__ float sInv;
  const int b = blockIdx.x;
  const int g = threadIdx.x;  // 128 threads

  const float* rg = r1 + ((size_t)b * G_ + g) * 20;
  float s = qb2[0];
#pragma unroll
  for (int j = 0; j < 10; ++j) {
    float u = qb1[j];
#pragma unroll
    for (int k = 0; k < 20; ++k) u += qW1[j * 20 + k] * rg[k];
    s += qW2[j] * fmaxf(u, 0.f);
  }
  sS[g] = s;
  __syncthreads();
  if (g == 0) {
    float mx = -1e30f;
    for (int i = 0; i < G_; ++i) mx = fmaxf(mx, sS[i]);
    float sum = 0.f;
    for (int i = 0; i < G_; ++i) { float e = __expf(sS[i] - mx); sS[i] = e; sum += e; }
    sInv = 1.f / sum;
  }
  __syncthreads();
  if (g < 20) {
    const float inv = sInv;
    float acc = 0.f;
    for (int i = 0; i < G_; ++i) acc += r1[((size_t)b * G_ + i) * 20 + g] * sS[i];
    sR[g] = acc * inv;
  }
  __syncthreads();
  if (g == 0) {
    float nrm = 0.f;
    for (int n = 0; n < 20; ++n) nrm += sR[n] * sR[n];
    nrm = fmaxf(sqrtf(nrm), 1e-12f);
    float rn[20];
    for (int n = 0; n < 20; ++n) rn[n] = sR[n] / nrm;
    float o = b3b[0];
#pragma unroll
    for (int j = 0; j < 10; ++j) {
      float u = b3a[j];
#pragma unroll
      for (int k = 0; k < 20; ++k) u += W3a[j * 20 + k] * rn[k];
      o += W3b[j] * fmaxf(u, 0.f);
    }
    out[b] = o;
  }
}

extern "C" void kernel_launch(void* const* d_in, const int* in_sizes, int n_in,
                              void* d_out, int out_size, void* d_ws, size_t ws_size,
                              hipStream_t stream) {
  const float* x   = (const float*)d_in[0];
  const float* W1a = (const float*)d_in[1];
  const float* b1a = (const float*)d_in[2];
  const float* W1b = (const float*)d_in[3];
  const float* b1b = (const float*)d_in[4];
  const float* W1c = (const float*)d_in[5];
  const float* b1c = (const float*)d_in[6];
  const float* pW1 = (const float*)d_in[7];
  const float* pb1 = (const float*)d_in[8];
  const float* pW2 = (const float*)d_in[9];
  const float* pb2 = (const float*)d_in[10];
  const float* qW1 = (const float*)d_in[11];
  const float* qb1 = (const float*)d_in[12];
  const float* qW2 = (const float*)d_in[13];
  const float* qb2 = (const float*)d_in[14];
  const float* W3a = (const float*)d_in[15];
  const float* b3a = (const float*)d_in[16];
  const float* W3b = (const float*)d_in[17];
  const float* b3b = (const float*)d_in[18];

  float* r1  = (float*)d_ws;   // B*G*20 f32 = 320 KB scratch
  float* out = (float*)d_out;  // (B,1) f32

  hipLaunchKernelGGL(fused_mlp_pool_t, dim3(B_ * G_), dim3(256), 0, stream,
                     x, W1a, b1a, W1b, b1b, W1c, b1c, pW1, pb1, pW2, pb2, r1);
  hipLaunchKernelGGL(pool_g_head, dim3(B_), dim3(128), 0, stream,
                     r1, qW1, qb1, qW2, qb2, W3a, b3a, W3b, b3b, out);
}
